// CrossAttentionBlock_58935541236230
// MI455X (gfx1250) — matrix operand order repaired and checked by reading, not compiled, not run on hardware
//
#include <hip/hip_runtime.h>
#include <hip/hip_bf16.h>
#include <stdint.h>

// ---- problem constants (fixed by the reference) ----
#define N_TOK 131072
#define BB    8
#define MM    256
#define DD    512
#define CC    768
#define HH    8
#define HD    64
#define NB    (N_TOK / BB)      // 16384 queries per batch
#define ATT_SCALE 0.125f        // 64^-0.5
#define EPS_LN 1e-5f

typedef __bf16        v16bf __attribute__((ext_vector_type(16)));
typedef float         v8f   __attribute__((ext_vector_type(8)));
typedef unsigned int  v4u   __attribute__((ext_vector_type(4)));
typedef float         v4f   __attribute__((ext_vector_type(4)));
typedef unsigned short u16;

union Frag { v16bf v; v4u u[2]; };

__device__ __forceinline__ u16 f2bf(float f) {
  union { float f; unsigned int u; } c; c.f = f;
  unsigned int lsb = (c.u >> 16) & 1u;               // round-to-nearest-even
  return (u16)((c.u + 0x7fffu + lsb) >> 16);
}

// ---- CDNA5 async-to-LDS path (ASYNCcnt-tracked DMA, ISA §10 / §15.18) ----
__device__ __forceinline__ void async_copy16(unsigned lds_off, const u16* gptr) {
  unsigned long long ga = (unsigned long long)(uintptr_t)gptr;
  asm volatile("global_load_async_to_lds_b128 %0, %1, off"
               :: "v"(lds_off), "v"(ga) : "memory");
}
__device__ __forceinline__ void wait_async_le4() {
  asm volatile("s_wait_asynccnt 0x4" ::: "memory");
}
__device__ __forceinline__ void wait_async_0() {
  asm volatile("s_wait_asynccnt 0x0" ::: "memory");
}

// ---------------------------------------------------------------------------
// LayerNorm: one wave (32 lanes) per 512-wide row; f32 out (residual) + bf16.
// ---------------------------------------------------------------------------
__global__ __launch_bounds__(256) void ln_kernel(
    const float* __restrict__ x, const float* __restrict__ g,
    const float* __restrict__ b, float* __restrict__ f32out,
    u16* __restrict__ bf16out) {
  int wave = threadIdx.x >> 5, lane = threadIdx.x & 31;
  int row = blockIdx.x * 8 + wave;
  const float* xr = x + (size_t)row * DD;
  float vals[16];
  float s = 0.f, ss = 0.f;
#pragma unroll
  for (int i = 0; i < 4; ++i) {
    v4f t = *(const v4f*)(xr + lane * 16 + i * 4);
#pragma unroll
    for (int j = 0; j < 4; ++j) { float v = t[j]; vals[i*4+j] = v; s += v; ss += v*v; }
  }
#pragma unroll
  for (int o = 16; o; o >>= 1) { s += __shfl_xor(s, o, 32); ss += __shfl_xor(ss, o, 32); }
  float mu   = s * (1.f / DD);
  float var  = ss * (1.f / DD) - mu * mu;
  float rstd = rsqrtf(var + EPS_LN);
#pragma unroll
  for (int i = 0; i < 16; ++i) {
    int col = lane * 16 + i;
    float y = (vals[i] - mu) * rstd * g[col] + b[col];
    f32out[(size_t)row * DD + col]  = y;
    bf16out[(size_t)row * DD + col] = f2bf(y);
  }
}

// ---------------------------------------------------------------------------
// Weight cast + transpose: Wt[c][k] = bf16(W[k][c])  (B-fragment friendly)
// ---------------------------------------------------------------------------
__global__ void wcvt_kernel(const float* __restrict__ W, u16* __restrict__ Wt,
                            int K, int Ncols) {
  int tid = blockIdx.x * blockDim.x + threadIdx.x;
  if (tid >= K * Ncols) return;
  int k = tid / Ncols, c = tid - k * Ncols;
  Wt[(size_t)c * K + k] = f2bf(W[tid]);
}

__global__ void cast_kernel(const float* __restrict__ in, u16* __restrict__ out, int n) {
  int tid = blockIdx.x * blockDim.x + threadIdx.x;
  if (tid < n) out[tid] = f2bf(in[tid]);
}

// ---------------------------------------------------------------------------
// WMMA GEMM: out[n_rows x n_cols] = X[n_rows x K] @ Wt^T + bias (+ epilogue)
//   X row-major bf16, Wt = transposed weight (n_cols x K) bf16.
//   Block = 8 waves; wave owns a 16x64 tile (4 chained v8f accumulators).
//   MODE 0: bf16 row-major   MODE 1: bf16 scatter to V^T (B,H,HD,M)
//   MODE 2: f32 + residual (final projection)
// ---------------------------------------------------------------------------
template <int MODE>
__global__ __launch_bounds__(256) void gemm_kernel(
    const u16* __restrict__ X, const u16* __restrict__ Wt,
    const float* __restrict__ bias, void* __restrict__ outp,
    const float* __restrict__ resid, int n_rows, int Kd, int n_cols) {
  int wave = threadIdx.x >> 5, lane = threadIdx.x & 31;
  int m0 = blockIdx.y * 128 + wave * 16;
  int n0 = blockIdx.x * 64;
  int lrow = lane & 15, hi = lane >> 4;

  v8f acc[4] = {};

  // A: lane half 0 holds K = {k0..k0+7, k0+16..k0+23}; half 1 is +8.
  const u16* xrow = X + (size_t)(m0 + lrow) * Kd + hi * 8;
  const u16* wcol[4];
#pragma unroll
  for (int j = 0; j < 4; ++j)
    wcol[j] = Wt + (size_t)(n0 + j * 16 + lrow) * Kd + hi * 8;

  for (int k0 = 0; k0 < Kd; k0 += 32) {
    __builtin_prefetch(xrow + k0 + 128, 0, 1);   // global_prefetch_b8
    Frag a;
    a.u[0] = *(const v4u*)(xrow + k0);
    a.u[1] = *(const v4u*)(xrow + k0 + 16);
#pragma unroll
    for (int j = 0; j < 4; ++j) {
      Frag bf;
      const u16* p = wcol[j] + k0;
      bf.u[0] = *(const v4u*)(p);
      bf.u[1] = *(const v4u*)(p + 16);
      acc[j] = __builtin_amdgcn_wmma_f32_16x16x32_bf16(
          false, a.v, false, bf.v, (short)0, acc[j], false, false);
    }
  }

#pragma unroll
  for (int j = 0; j < 4; ++j) {
    int col = n0 + j * 16 + lrow;
    float bv = bias[col];
#pragma unroll
    for (int r = 0; r < 8; ++r) {
      int row = m0 + r + hi * 8;                // C/D layout: VGPR r -> M=r / M=r+8
      float y = acc[j][r] + bv;
      if (MODE == 0) {
        ((u16*)outp)[(size_t)row * n_cols + col] = f2bf(y);
      } else if (MODE == 1) {
        int bi = row >> 8, mi = row & 255;      // row = b*256 + m
        int hh = col >> 6, hd = col & 63;       // col = h*64 + hd
        ((u16*)outp)[((size_t)(bi * HH + hh) * HD + hd) * MM + mi] = f2bf(y);
      } else {
        size_t idx = (size_t)row * n_cols + col;
        ((float*)outp)[idx] = y + resid[idx];
      }
    }
  }
}

// ---------------------------------------------------------------------------
// Attention: one wave per (head, 16-query tile). M=256 keys fit in one pass,
// so full softmax (no online rescale).
//   - Q A-fragments register-resident.
//   - K tiles streamed through a per-wave LDS double buffer with
//     GLOBAL_LOAD_ASYNC_TO_LDS_B128 (ASYNCcnt), staged one tile ahead so the
//     QK^T WMMAs never wait on global latency (K/V are L2-resident: 4 MB
//     inside the 192 MB L2).
//   - S in LDS (f32), P in LDS (bf16), O = P @ V^T reads A straight from LDS.
// ---------------------------------------------------------------------------
__global__ __launch_bounds__(64) void attn_kernel(
    const u16* __restrict__ q, const u16* __restrict__ kk,
    const u16* __restrict__ vT, u16* __restrict__ attended) {
  __shared__ float sS[2][16 * 256];        // 32 KB
  __shared__ u16   sP[2][16 * 256];        // 16 KB
  __shared__ u16   sK[2][2][16 * HD];      //  8 KB: per-wave K double buffer
  int wave = threadIdx.x >> 5, lane = threadIdx.x & 31;
  int w    = blockIdx.x * 2 + wave;
  int h    = w >> 13;                 // 8192 query-tiles per head
  int tile = w & 8191;
  int row0 = tile << 4;
  int bidx = row0 >> 14;              // NB = 16384 queries per batch
  int lrow = lane & 15, hi = lane >> 4;

  // Q A-fragments (K = 64 -> two 16x32 fragments), resident for whole kernel.
  const u16* qrow = q + (size_t)(row0 + lrow) * DD + h * HD + hi * 8;
  Frag aq0, aq1;
  aq0.u[0] = *(const v4u*)(qrow + 0);
  aq0.u[1] = *(const v4u*)(qrow + 16);
  aq1.u[0] = *(const v4u*)(qrow + 32);
  aq1.u[1] = *(const v4u*)(qrow + 48);

  float* Srow = sS[wave];
  u16*   Prow = sP[wave];
  unsigned ldsK[2];
  ldsK[0] = (unsigned)(uintptr_t)&sK[wave][0][0];
  ldsK[1] = (unsigned)(uintptr_t)&sK[wave][1][0];

  // Stage one 16-key x 64-hd tile (2 KB) = 4 async b128 copies per wave.
  const u16* kbase = kk + (size_t)(bidx * MM) * DD + h * HD;
  auto stage_k = [&](int buf, int kt) {
#pragma unroll
    for (int i = 0; i < 4; ++i) {
      int cid = i * 32 + lane;            // 128 x 16B chunks per tile
      int m   = cid >> 3;                 // key within tile (8 chunks / row)
      int co  = (cid & 7) * 8;            // hd offset (u16)
      async_copy16(ldsK[buf] + cid * 16,
                   kbase + (size_t)(kt * 16 + m) * DD + co);
    }
  };

  // ---- S = (Q K^T) * scale : async-pipelined over 16 key tiles ----
  stage_k(0, 0);
  for (int kt = 0; kt < 16; ++kt) {
    int buf = kt & 1;
    if (kt < 15) { stage_k(buf ^ 1, kt + 1); wait_async_le4(); }
    else         { wait_async_0(); }
    const u16* kp = &sK[wave][buf][0] + lrow * HD + hi * 8;   // ds_load_b128
    Frag b0, b1;
    b0.u[0] = *(const v4u*)(kp);      b0.u[1] = *(const v4u*)(kp + 16);
    b1.u[0] = *(const v4u*)(kp + 32); b1.u[1] = *(const v4u*)(kp + 48);
    v8f s = {};
    s = __builtin_amdgcn_wmma_f32_16x16x32_bf16(false, aq0.v, false, b0.v, (short)0, s, false, false);
    s = __builtin_amdgcn_wmma_f32_16x16x32_bf16(false, aq1.v, false, b1.v, (short)0, s, false, false);
#pragma unroll
    for (int r = 0; r < 8; ++r)
      Srow[(r + hi * 8) * 256 + kt * 16 + lrow] = s[r] * ATT_SCALE;
  }

  // ---- softmax over 256 keys, per row; wave-wide shfl reductions ----
#pragma unroll 1
  for (int r = 0; r < 16; ++r) {
    float x[8];
#pragma unroll
    for (int i = 0; i < 8; ++i) x[i] = Srow[r * 256 + lane * 8 + i];
    float mx = x[0];
#pragma unroll
    for (int i = 1; i < 8; ++i) mx = fmaxf(mx, x[i]);
#pragma unroll
    for (int o = 16; o; o >>= 1) mx = fmaxf(mx, __shfl_xor(mx, o, 32));
    float sum = 0.f;
#pragma unroll
    for (int i = 0; i < 8; ++i) { x[i] = __expf(x[i] - mx); sum += x[i]; }
#pragma unroll
    for (int o = 16; o; o >>= 1) sum += __shfl_xor(sum, o, 32);
    float inv = 1.f / sum;
#pragma unroll
    for (int i = 0; i < 8; ++i)
      Prow[r * 256 + lane * 8 + i] = f2bf(x[i] * inv);
  }

  // ---- O = P @ V : 4 HD column tiles, K = 256 in chunks of 32 ----
#pragma unroll 1
  for (int ct = 0; ct < 4; ++ct) {
    v8f o = {};
    const u16* vp = vT + ((size_t)((bidx * HH + h) * HD + ct * 16 + lrow)) * MM + hi * 8;
    __builtin_prefetch(vp, 0, 1);
    for (int kc = 0; kc < 8; ++kc) {
      Frag a, bf;
      const u16* pp = Prow + lrow * 256 + kc * 32 + hi * 8;   // LDS (ds_load)
      a.u[0]  = *(const v4u*)(pp);
      a.u[1]  = *(const v4u*)(pp + 16);
      bf.u[0] = *(const v4u*)(vp + kc * 32);
      bf.u[1] = *(const v4u*)(vp + kc * 32 + 16);
      o = __builtin_amdgcn_wmma_f32_16x16x32_bf16(
          false, a.v, false, bf.v, (short)0, o, false, false);
    }
#pragma unroll
    for (int r = 0; r < 8; ++r)
      attended[(size_t)(row0 + r + hi * 8) * DD + h * HD + ct * 16 + lrow] = f2bf(o[r]);
  }
}

// ---------------------------------------------------------------------------
extern "C" void kernel_launch(void* const* d_in, const int* in_sizes, int n_in,
                              void* d_out, int out_size, void* d_ws, size_t ws_size,
                              hipStream_t stream) {
  (void)in_sizes; (void)n_in; (void)out_size; (void)ws_size;
  const float* features = (const float*)d_in[0];
  // d_in[1] = batch_indices: sorted, equal counts -> implicit in index math
  const float* context  = (const float*)d_in[2];
  const float* Wq   = (const float*)d_in[3];
  const float* bq   = (const float*)d_in[4];
  const float* Wk   = (const float*)d_in[5];
  const float* bk   = (const float*)d_in[6];
  const float* Wv   = (const float*)d_in[7];
  const float* bv   = (const float*)d_in[8];
  const float* Wout = (const float*)d_in[9];
  const float* bout = (const float*)d_in[10];
  const float* ln_g = (const float*)d_in[11];
  const float* ln_b = (const float*)d_in[12];
  float* out = (float*)d_out;

  char* ws = (char*)d_ws;
  size_t o = 0;
  float* fbuf = (float*)(ws + o); o += (size_t)N_TOK * DD * 4;   // LN f32 (residual)
  u16* f16   = (u16*)(ws + o);    o += (size_t)N_TOK * DD * 2;   // LN bf16
  u16* qb    = (u16*)(ws + o);    o += (size_t)N_TOK * DD * 2;   // Q bf16
  u16* att   = (u16*)(ws + o);    o += (size_t)N_TOK * DD * 2;   // attended bf16
  u16* kb    = (u16*)(ws + o);    o += (size_t)BB * MM * DD * 2; // K bf16 (B*M, D)
  u16* vt    = (u16*)(ws + o);    o += (size_t)BB * MM * DD * 2; // V^T bf16 (B,H,HD,M)
  u16* wqT   = (u16*)(ws + o);    o += (size_t)DD * DD * 2;
  u16* wkT   = (u16*)(ws + o);    o += (size_t)CC * DD * 2;
  u16* wvT   = (u16*)(ws + o);    o += (size_t)CC * DD * 2;
  u16* woT   = (u16*)(ws + o);    o += (size_t)DD * DD * 2;
  u16* ctx16 = (u16*)(ws + o);    o += (size_t)BB * MM * CC * 2;

  ln_kernel<<<N_TOK / 8, 256, 0, stream>>>(features, ln_g, ln_b, fbuf, f16);
  wcvt_kernel<<<(DD * DD + 255) / 256, 256, 0, stream>>>(Wq, wqT, DD, DD);
  wcvt_kernel<<<(CC * DD + 255) / 256, 256, 0, stream>>>(Wk, wkT, CC, DD);
  wcvt_kernel<<<(CC * DD + 255) / 256, 256, 0, stream>>>(Wv, wvT, CC, DD);
  wcvt_kernel<<<(DD * DD + 255) / 256, 256, 0, stream>>>(Wout, woT, DD, DD);
  cast_kernel<<<(BB * MM * CC + 255) / 256, 256, 0, stream>>>(context, ctx16, BB * MM * CC);

  // Q = LN(x) @ Wq + bq
  gemm_kernel<0><<<dim3(DD / 64, N_TOK / 128), 256, 0, stream>>>(
      f16, wqT, bq, qb, nullptr, N_TOK, DD, DD);
  // K = ctx @ Wk + bk
  gemm_kernel<0><<<dim3(DD / 64, (BB * MM) / 128), 256, 0, stream>>>(
      ctx16, wkT, bk, kb, nullptr, BB * MM, CC, DD);
  // V = ctx @ Wv + bv, stored transposed (B,H,HD,M)
  gemm_kernel<1><<<dim3(DD / 64, (BB * MM) / 128), 256, 0, stream>>>(
      ctx16, wvT, bv, vt, nullptr, BB * MM, CC, DD);

  // softmax(Q K^T * scale) @ V
  attn_kernel<<<(N_TOK / 16) * HH / 2, 64, 0, stream>>>(qb, kb, vt, att);

  // out = attended @ Wout + bout + LN(x)
  gemm_kernel<2><<<dim3(DD / 64, N_TOK / 128), 256, 0, stream>>>(
      att, woT, bout, out, fbuf, N_TOK, DD, DD);
}
